// GASTON_MoE_76218489635144
// MI455X (gfx1250) — matrix operand order repaired and checked by reading, not compiled, not run on hardware
//
#include <hip/hip_runtime.h>
#include <hip/hip_bf16.h>

#define NPTS 65536
#define GG   1024
#define EE   8
#define SHD  64
#define AHD  256
#define GHD  64
#define MBLK 64   // points per expert block (4 x 16-row WMMA subtiles)

typedef __attribute__((ext_vector_type(16))) _Float16 v16h;
typedef __attribute__((ext_vector_type(8)))  float    v8f;

union Frag16 { v16h h; uint4 q[2]; };

// ---- WMMA fragment loaders (wave32, 16x16x32 f16) -------------------------
// A matrix 16x32 (MxK), row-major source with leading dim `ld` (f16 elems):
// lane<16 holds K {0..7, 16..23}, lane>=16 holds K {8..15, 24..31}; row = lane&15.
__device__ inline v16h load_fragA(const _Float16* src, int ld, int k0, int lane) {
    int m  = lane & 15;
    int kh = (lane < 16) ? 0 : 8;
    Frag16 f;
    f.q[0] = *(const uint4*)(const void*)(src + m * ld + k0 + kh);
    f.q[1] = *(const uint4*)(const void*)(src + m * ld + k0 + 16 + kh);
    return f.h;
}

// B matrix 32x16 (KxN) from row-major W[out][in] (out = N dim, in = K dim):
// lane<16 holds K {0..7, 16..23}, lane>=16 holds K {8..15, 24..31}; col = lane&15.
__device__ inline v16h load_fragB(const _Float16* w, int ldk, int ntile, int k0, int lane) {
    int n  = lane & 15;
    int kh = (lane < 16) ? 0 : 8;
    const _Float16* p = w + (size_t)(ntile * 16 + n) * ldk + k0 + kh;
    Frag16 f;
    f.q[0] = *(const uint4*)(const void*)(p);
    f.q[1] = *(const uint4*)(const void*)(p + 16);
    return f.h;
}

// ---- weight f32 -> f16 conversion -----------------------------------------
__global__ void cvt_f16_kernel(const float* __restrict__ sW2,
                               const float* __restrict__ aW2,
                               const float* __restrict__ aW3,
                               _Float16* __restrict__ dS,
                               _Float16* __restrict__ dA2,
                               _Float16* __restrict__ dA3) {
    const int nS  = EE * SHD * SHD;
    const int nA2 = EE * AHD * AHD;
    const int nA3 = EE * GG * AHD;
    const int total = nS + nA2 + nA3;
    for (int i = blockIdx.x * blockDim.x + threadIdx.x; i < total;
         i += gridDim.x * blockDim.x) {
        if (i < nS)            dS[i]             = (_Float16)sW2[i];
        else if (i < nS + nA2) dA2[i - nS]       = (_Float16)aW2[i - nS];
        else                   dA3[i - nS - nA2] = (_Float16)aW3[i - nS - nA2];
    }
}

// ---- gate MLP + top-2 dispatch --------------------------------------------
__global__ __launch_bounds__(256)
void gate_kernel(const float* __restrict__ x,
                 const float* __restrict__ gW1, const float* __restrict__ gb1,
                 const float* __restrict__ gW2, const float* __restrict__ gb2,
                 const float* __restrict__ gW3, const float* __restrict__ gb3,
                 int* __restrict__ cnt, float* __restrict__ imp,
                 int* __restrict__ lidx, float* __restrict__ lgate) {
    __shared__ float sW1[GHD * 2], sb1[GHD];
    __shared__ float sW2[GHD * GHD], sb2[GHD];
    __shared__ float sW3[EE * GHD], sb3[EE];
    int tid = threadIdx.x;
    for (int i = tid; i < GHD * 2; i += 256) sW1[i] = gW1[i];
    for (int i = tid; i < GHD; i += 256)     { sb1[i] = gb1[i]; sb2[i] = gb2[i]; }
    for (int i = tid; i < GHD * GHD; i += 256) sW2[i] = gW2[i];
    for (int i = tid; i < EE * GHD; i += 256)  sW3[i] = gW3[i];
    if (tid < EE) sb3[tid] = gb3[tid];
    __syncthreads();

    int n = blockIdx.x * 256 + tid;
    float x0 = x[2 * n], x1 = x[2 * n + 1];

    float h1[GHD];
#pragma unroll
    for (int j = 0; j < GHD; ++j)
        h1[j] = fmaxf(sW1[j * 2] * x0 + sW1[j * 2 + 1] * x1 + sb1[j], 0.f);
    float h2[GHD];
    for (int j = 0; j < GHD; ++j) {
        float s = sb2[j];
#pragma unroll
        for (int i = 0; i < GHD; ++i) s += sW2[j * GHD + i] * h1[i];
        h2[j] = fmaxf(s, 0.f);
    }
    float lg[EE];
    for (int e = 0; e < EE; ++e) {
        float s = sb3[e];
#pragma unroll
        for (int i = 0; i < GHD; ++i) s += sW3[e * GHD + i] * h2[i];
        lg[e] = s;
    }
    // top-2 (first max on ties, matching jax top_k stability)
    float b1v = -1e30f; int i0 = 0;
    for (int e = 0; e < EE; ++e) if (lg[e] > b1v) { b1v = lg[e]; i0 = e; }
    float b2v = -1e30f; int i1 = 0;
    for (int e = 0; e < EE; ++e) if (e != i0 && lg[e] > b2v) { b2v = lg[e]; i1 = e; }
    float d = __expf(b2v - b1v);
    float inv = 1.f / (1.f + d);
    float g0 = inv, g1 = d * inv;

    atomicAdd(&imp[i0], g0);
    atomicAdd(&imp[i1], g1);
    int p0 = atomicAdd(&cnt[i0], 1);
    lidx[i0 * NPTS + p0] = n; lgate[i0 * NPTS + p0] = g0;
    int p1 = atomicAdd(&cnt[i1], 1);
    lidx[i1 * NPTS + p1] = n; lgate[i1 * NPTS + p1] = g1;
}

// ---- per-(expert, 64-point tile) expert evaluation, WMMA core -------------
// M is register-blocked 4x: each weight (B) fragment feeds 4 WMMAs.
__global__ __launch_bounds__(256)
void expert_kernel(const float* __restrict__ x,
                   const float* __restrict__ S_W1, const float* __restrict__ S_b1,
                   const float* __restrict__ S_b2,
                   const float* __restrict__ S_W3, const float* __restrict__ S_b3,
                   const float* __restrict__ A_W1, const float* __restrict__ A_b1,
                   const float* __restrict__ A_b2, const float* __restrict__ A_b3,
                   const _Float16* __restrict__ S_W2h,
                   const _Float16* __restrict__ A_W2h,
                   const _Float16* __restrict__ A_W3h,
                   const int* __restrict__ cnt,
                   const int* __restrict__ lidx, const float* __restrict__ lgate,
                   float* __restrict__ out) {
    const int e = blockIdx.y;
    const int count = cnt[e];
    const int tile = blockIdx.x;
    if (tile * MBLK >= count) return;

    __shared__ int   s_pt[MBLK];
    __shared__ float s_g[MBLK];
    __shared__ float s_x0[MBLK], s_x1[MBLK];
    __shared__ float s_iso[MBLK];
    __shared__ __align__(16) _Float16 sh1[MBLK * 72];   // S hidden1 (64 + pad)
    __shared__ __align__(16) _Float16 sh2[MBLK * 72];   // S hidden2
    __shared__ __align__(16) _Float16 ah1[MBLK * 264];  // A hidden1 (256 + pad)
    __shared__ __align__(16) _Float16 ah2[MBLK * 264];  // A hidden2

    const int tid  = threadIdx.x;
    const int lane = tid & 31;
    const int wave = tid >> 5;
    const int mhi  = (lane < 16) ? 0 : 8;
    const int col  = lane & 15;

    for (int m = tid; m < MBLK; m += 256) {
        int r = tile * MBLK + m;
        if (r < count) {
            int pt = lidx[e * NPTS + r];
            s_pt[m] = pt;
            s_g[m]  = lgate[e * NPTS + r];
            s_x0[m] = x[2 * pt];
            s_x1[m] = x[2 * pt + 1];
        } else {
            s_pt[m] = -1; s_g[m] = 0.f; s_x0[m] = 0.f; s_x1[m] = 0.f;
        }
    }
    __syncthreads();

    // S-net layer1: 2 -> 64 (scalar, tiny K)
    for (int idx = tid; idx < MBLK * SHD; idx += 256) {
        int m = idx >> 6, j = idx & 63;
        float v = s_x0[m] * S_W1[e * SHD * 2 + j * 2] +
                  s_x1[m] * S_W1[e * SHD * 2 + j * 2 + 1] + S_b1[e * SHD + j];
        sh1[m * 72 + j] = (_Float16)fmaxf(v, 0.f);
    }
    __syncthreads();

    // S-net layer2: 64 -> 64 via WMMA. D is 64x64: 4 msub x 4 ntile = 16 jobs.
    {
        const _Float16* w = S_W2h + e * SHD * SHD;
        for (int job = wave; job < 16; job += 8) {
            int ms = job >> 2, nt = job & 3;
            v8f acc = {};
#pragma unroll
            for (int kb = 0; kb < 2; ++kb) {
                v16h a = load_fragA(sh1 + ms * 16 * 72, 72, kb * 32, lane);
                v16h b = load_fragB(w, SHD, nt, kb * 32, lane);
                acc = __builtin_amdgcn_wmma_f32_16x16x32_f16(false, a, false, b,
                                                             (short)0, acc, false, false);
            }
            int nn = nt * 16 + col;
            float bias = S_b2[e * SHD + nn];
#pragma unroll
            for (int r = 0; r < 8; ++r)
                sh2[(ms * 16 + r + mhi) * 72 + nn] = (_Float16)fmaxf(acc[r] + bias, 0.f);
        }
    }
    __syncthreads();

    // S-net layer3: 64 -> 1 (isodepth, scalar)
    for (int m = tid; m < MBLK; m += 256) {
        float s = S_b3[e];
#pragma unroll
        for (int j = 0; j < SHD; ++j)
            s += (float)sh2[m * 72 + j] * S_W3[e * SHD + j];
        s_iso[m] = s;
    }
    __syncthreads();

    // A-net layer1: 1 -> 256 (outer product, scalar)
    for (int idx = tid; idx < MBLK * AHD; idx += 256) {
        int m = idx >> 8, j = idx & 255;
        float v = s_iso[m] * A_W1[e * AHD + j] + A_b1[e * AHD + j];
        ah1[m * 264 + j] = (_Float16)fmaxf(v, 0.f);
    }
    __syncthreads();

    // A-net layer2: 256 -> 256 via WMMA. 16 ntiles, 2 per wave,
    // 4-way M register blocking: 1 B-frag load feeds 4 WMMAs.
    {
        const _Float16* w = A_W2h + (size_t)e * AHD * AHD;
        for (int t = wave; t < 16; t += 8) {
            v8f acc[4] = {{}, {}, {}, {}};
#pragma unroll
            for (int kb = 0; kb < 8; ++kb) {
                v16h b = load_fragB(w, AHD, t, kb * 32, lane);
#pragma unroll
                for (int ms = 0; ms < 4; ++ms) {
                    v16h a = load_fragA(ah1 + ms * 16 * 264, 264, kb * 32, lane);
                    acc[ms] = __builtin_amdgcn_wmma_f32_16x16x32_f16(
                        false, a, false, b, (short)0, acc[ms], false, false);
                }
            }
            int nn = t * 16 + col;
            float bias = A_b2[e * AHD + nn];
#pragma unroll
            for (int ms = 0; ms < 4; ++ms)
#pragma unroll
                for (int r = 0; r < 8; ++r)
                    ah2[(ms * 16 + r + mhi) * 264 + nn] =
                        (_Float16)fmaxf(acc[ms][r] + bias, 0.f);
        }
    }
    __syncthreads();

    // A-net layer3: 256 -> 1024 via WMMA. 64 ntiles, 8 per wave,
    // 4-way M register blocking, then gate-weighted atomic combine
    // (exactly 2 commutative f32 adds per output element -> deterministic).
    {
        const _Float16* w = A_W3h + (size_t)e * GG * AHD;
        for (int t = wave * 8; t < wave * 8 + 8; ++t) {
            v8f acc[4] = {{}, {}, {}, {}};
#pragma unroll
            for (int kb = 0; kb < 8; ++kb) {
                v16h b = load_fragB(w, AHD, t, kb * 32, lane);
#pragma unroll
                for (int ms = 0; ms < 4; ++ms) {
                    v16h a = load_fragA(ah2 + ms * 16 * 264, 264, kb * 32, lane);
                    acc[ms] = __builtin_amdgcn_wmma_f32_16x16x32_f16(
                        false, a, false, b, (short)0, acc[ms], false, false);
                }
            }
            int nn = t * 16 + col;
            float bias = A_b3[e * GG + nn];
#pragma unroll
            for (int ms = 0; ms < 4; ++ms) {
#pragma unroll
                for (int r = 0; r < 8; ++r) {
                    int m = ms * 16 + r + mhi;
                    int pt = s_pt[m];
                    if (pt >= 0)
                        atomicAdd(out + (size_t)pt * GG + nn,
                                  s_g[m] * (acc[ms][r] + bias));
                }
            }
        }
    }
}

// ---- load-balancing loss ---------------------------------------------------
__global__ void loss_kernel(const float* __restrict__ imp,
                            const int* __restrict__ cnt,
                            float* __restrict__ loss_out) {
    float vi[EE], vl[EE];
    float mi = 0.f, ml = 0.f;
    for (int e = 0; e < EE; ++e) {
        vi[e] = imp[e]; vl[e] = (float)cnt[e];
        mi += vi[e]; ml += vl[e];
    }
    mi /= EE; ml /= EE;
    float si = 0.f, sl = 0.f;
    for (int e = 0; e < EE; ++e) {
        si += (vi[e] - mi) * (vi[e] - mi);
        sl += (vl[e] - ml) * (vl[e] - ml);
    }
    si /= (EE - 1); sl /= (EE - 1);
    const float eps = 1e-10f;
    float cv2i = si / (mi * mi + eps);
    float cv2l = sl / (ml * ml + eps);
    *loss_out = (cv2i + cv2l) * 0.01f;
}

extern "C" void kernel_launch(void* const* d_in, const int* in_sizes, int n_in,
                              void* d_out, int out_size, void* d_ws, size_t ws_size,
                              hipStream_t stream) {
    const float* x    = (const float*)d_in[0];
    const float* S_W1 = (const float*)d_in[1];
    const float* S_b1 = (const float*)d_in[2];
    const float* S_W2 = (const float*)d_in[3];
    const float* S_b2 = (const float*)d_in[4];
    const float* S_W3 = (const float*)d_in[5];
    const float* S_b3 = (const float*)d_in[6];
    const float* A_W1 = (const float*)d_in[7];
    const float* A_b1 = (const float*)d_in[8];
    const float* A_W2 = (const float*)d_in[9];
    const float* A_b2 = (const float*)d_in[10];
    const float* A_W3 = (const float*)d_in[11];
    const float* A_b3 = (const float*)d_in[12];
    const float* g_W1 = (const float*)d_in[13];
    const float* g_b1 = (const float*)d_in[14];
    const float* g_W2 = (const float*)d_in[15];
    const float* g_b2 = (const float*)d_in[16];
    const float* g_W3 = (const float*)d_in[17];
    const float* g_b3 = (const float*)d_in[18];

    float* out = (float*)d_out;

    // workspace layout
    char* ws = (char*)d_ws;
    int*   cnt   = (int*)ws;                                   // 8 ints
    float* imp   = (float*)(ws + 32);                          // 8 floats
    int*   lidx  = (int*)(ws + 256);                           // EE*NPTS ints
    float* lgate = (float*)(ws + 256 + (size_t)EE * NPTS * 4); // EE*NPTS floats
    _Float16* S_W2h = (_Float16*)(ws + 256 + (size_t)EE * NPTS * 8);
    _Float16* A_W2h = S_W2h + EE * SHD * SHD;
    _Float16* A_W3h = A_W2h + (size_t)EE * AHD * AHD;

    hipMemsetAsync(ws, 0, 64, stream);                               // cnt + imp
    hipMemsetAsync(d_out, 0, (size_t)NPTS * GG * sizeof(float), stream);

    cvt_f16_kernel<<<1024, 256, 0, stream>>>(S_W2, A_W2, A_W3, S_W2h, A_W2h, A_W3h);

    gate_kernel<<<NPTS / 256, 256, 0, stream>>>(x, g_W1, g_b1, g_W2, g_b2, g_W3, g_b3,
                                                cnt, imp, lidx, lgate);

    dim3 grid(NPTS / MBLK, EE);
    expert_kernel<<<grid, 256, 0, stream>>>(x, S_W1, S_b1, S_b2, S_W3, S_b3,
                                            A_W1, A_b1, A_b2, A_b3,
                                            S_W2h, A_W2h, A_W3h,
                                            cnt, lidx, lgate, out);

    loss_kernel<<<1, 1, 0, stream>>>(imp, cnt, out + (size_t)NPTS * GG);
}
